// DeepUnrecordedEGNN_44796508897959
// MI455X (gfx1250) — matrix operand order repaired and checked by reading, not compiled, not run on hardware
//
#include <hip/hip_runtime.h>
#include <hip/hip_bf16.h>
#include <stdint.h>

#define NN      50000
#define EE      800000
#define D_IN    128
#define DHID    64
#define NCH     4
#define NLAY    3
#define CLAMP_V 2.0f

// ---- CDNA5 WMMA types (wave32) -------------------------------------------
typedef __attribute__((ext_vector_type(16))) __bf16 v16bf;
typedef __attribute__((ext_vector_type(8)))  float  v8f;

// Native hardware convert (let the backend pick the bf16 cvt op).
__device__ __forceinline__ __bf16 f2bf(float f) { return (__bf16)f; }

__device__ __forceinline__ float sigmoidf_(float x) { return 1.0f / (1.0f + __expf(-x)); }
__device__ __forceinline__ float siluf_(float x)    { return x / (1.0f + __expf(-x)); }

__device__ __forceinline__ v8f wmma_bf(v16bf a, v16bf b, v8f c) {
  // D(16x16,f32) = A(16x32,bf16) x B(32x16,bf16) + C
  return __builtin_amdgcn_wmma_f32_16x16x32_bf16(false, a, false, b, (short)0, c, false, false);
}

// B fragment from pre-transposed weights Wt[Nout][K]:
// lane holds column n = lane&15; lanes 0-15 hold K {0..7, 16..23}, lanes 16-31 K {8..15, 24..31} (+kbase).
// Two 16B chunks per lane -> b128-class loads (global or ds).
__device__ __forceinline__ v16bf bfragT(const __bf16* Wt, int K, int nbase, int kbase) {
  const int lane = threadIdx.x & 31;
  const __bf16* p = Wt + (size_t)(nbase + (lane & 15)) * K + (kbase + ((lane >> 4) & 1) * 8);
  v16bf b; __builtin_memcpy(&b, p, 16); __builtin_memcpy((char*)&b + 16, p + 16, 16);
  return b;
}

// A fragment (16x32 bf16) from per-lane row pointer (row m = lane&15):
// lanes 0-15: K {0..7, 16..23}; lanes 16-31: K {8..15, 24..31} (ISA 7.12.2).
__device__ __forceinline__ v16bf afrag_bf(const __bf16* row, int kbase) {
  const int hi = (threadIdx.x >> 4) & 1;
  const __bf16* p = row + kbase + 8 * hi;
  v16bf a;
  __builtin_memcpy(&a, p, 16);
  __builtin_memcpy((char*)&a + 16, p + 16, 16);
  return a;
}

// Same A fragment, built from fp32 with on-the-fly bf16 conversion.
__device__ __forceinline__ v16bf afrag_f32(const float* row, int kbase) {
  const int hi = (threadIdx.x >> 4) & 1;
  const float* p = row + kbase + 8 * hi;
  v16bf a;
#pragma unroll
  for (int t = 0; t < 8; ++t) { a[t] = f2bf(p[t]); a[8 + t] = f2bf(p[16 + t]); }
  return a;
}

// ---- bf16 weight staging layout in workspace (elements) ------------------
//   [0, 32768)          mix_in^T  [256][128]
//   [32768, 65536)      mix_out^T [128][256]
//   65536 + c*24576:    edge-contiguous block: ew1_src^T[64][64], +4096 ew1_dst^T[64][64],
//                       +8192 ew2^T[64][64], +12288 pw1^T[64][64]   (16384 elems = 32KB)
//                       +16384 nw^T[64][128]                         (node kernel only)
#define WT_MIXOUT 32768
#define WT_CH     65536
#define WT_CH_STR 24576
#define WT_EDGE_ELEMS 16384
#define WT_NW_OFF 16384
#define WT_TOTAL  163840

#define STG_LD 72   // bf16 stage row stride: 144B, keeps 16B alignment for b128 ds loads

__global__ void prep_weights_kernel(const float* __restrict__ mixin,
                                    const float* __restrict__ mixout,
                                    const float* __restrict__ ew1,
                                    const float* __restrict__ ew2,
                                    const float* __restrict__ nw,
                                    const float* __restrict__ pw1,
                                    __bf16* __restrict__ wt) {
  const int tid = blockIdx.x * blockDim.x + threadIdx.x;
  const int stp = gridDim.x * blockDim.x;
  for (int i = tid; i < 256 * 128; i += stp) { int n = i >> 7, k = i & 127; wt[i] = f2bf(mixin[k * 256 + n]); }
  for (int i = tid; i < 128 * 256; i += stp) { int n = i >> 8, k = i & 255; wt[WT_MIXOUT + i] = f2bf(mixout[k * 128 + n]); }
  for (int c = 0; c < NCH; ++c) {
    const float* e1  = ew1 + (size_t)c * 129 * DHID;
    const float* e2  = ew2 + (size_t)c * DHID * DHID;
    const float* p1  = pw1 + (size_t)c * DHID * DHID;
    const float* nwc = nw  + (size_t)c * 2 * DHID * DHID;
    __bf16* o = wt + WT_CH + c * WT_CH_STR;
    for (int i = tid; i < 4096; i += stp) {
      int n = i >> 6, k = i & 63;
      o[i]         = f2bf(e1[k * 64 + n]);          // rows 0..63   (h_src part)
      o[4096 + i]  = f2bf(e1[(64 + k) * 64 + n]);   // rows 64..127 (h_dst part)
      o[8192 + i]  = f2bf(e2[k * 64 + n]);
      o[12288 + i] = f2bf(p1[k * 64 + n]);
    }
    for (int i = tid; i < 8192; i += stp) {
      int n = i >> 7, k = i & 127;
      o[WT_NW_OFF + i] = f2bf(nwc[k * 64 + n]);
    }
  }
}

// ---- small utility kernels ------------------------------------------------
__global__ void init_kernel(const float* __restrict__ x, const float* __restrict__ pos,
                            float* __restrict__ ox, float* __restrict__ op,
                            float* __restrict__ cnt) {
  const int i = blockIdx.x * blockDim.x + threadIdx.x;
  const int s = gridDim.x * blockDim.x;
  for (int j = i; j < NN * D_IN; j += s) ox[j] = x[j];
  for (int j = i; j < NN * 3; j += s) op[j] = pos[j];
  for (int j = i; j < NN; j += s) cnt[j] = 0.0f;
}

__global__ void count_kernel(const int* __restrict__ dst, float* __restrict__ cnt) {
  const int e = blockIdx.x * blockDim.x + threadIdx.x;
  if (e < EE) atomicAdd(&cnt[dst[e]], 1.0f);
}

__global__ void inv_kernel(float* __restrict__ cnt) {
  const int n = blockIdx.x * blockDim.x + threadIdx.x;
  if (n < NN) cnt[n] = 1.0f / ((float)NCH * fmaxf(cnt[n], 1.0f));
}

__global__ void zero_kernel(float* __restrict__ agg, float* __restrict__ possum) {
  const int i = blockIdx.x * blockDim.x + threadIdx.x;
  const int s = gridDim.x * blockDim.x;
  for (int j = i; j < NCH * NN * DHID; j += s) agg[j] = 0.0f;
  for (int j = i; j < NN * 3; j += s) possum[j] = 0.0f;
}

__global__ void pos_update_kernel(float* __restrict__ pos, const float* __restrict__ possum,
                                  const float* __restrict__ cntinv) {
  const int i = blockIdx.x * blockDim.x + threadIdx.x;
  if (i < NN * 3) pos[i] += possum[i] * cntinv[i / 3];
}

// ---- mix_in: u_bf16[N,256] = bf16( x[N,128] @ W[128,256] ) ---------------
__global__ __launch_bounds__(256)
void egnn_mixin_kernel(const float* __restrict__ x, const __bf16* __restrict__ wt,
                       __bf16* __restrict__ u_bf) {
  const int wave = threadIdx.x >> 5, lane = threadIdx.x & 31;
  const int ln = lane & 15, hi = lane >> 4;
  const int mtile = blockIdx.x;
  const int nbase = wave * 32;   // 8 waves x 32 cols = 256
  const float* row = x + (size_t)(mtile * 16 + ln) * D_IN;
  v16bf af[4];
#pragma unroll
  for (int kk = 0; kk < 4; ++kk) af[kk] = afrag_f32(row, kk * 32);
#pragma unroll
  for (int half = 0; half < 2; ++half) {
    v8f a = {};
#pragma unroll
    for (int kk = 0; kk < 4; ++kk)
      a = wmma_bf(af[kk], bfragT(wt, D_IN, nbase + half * 16, kk * 32), a);
#pragma unroll
    for (int v = 0; v < 8; ++v)
      u_bf[(size_t)(mtile * 16 + 8 * hi + v) * (NCH * DHID) + nbase + half * 16 + ln] = f2bf(a[v]);
  }
}

// ---- fused edge kernel: gather -> 3 WMMA GEMMs -> gate -> scatters -------
__global__ __launch_bounds__(256)
void egnn_edge_kernel(const __bf16* __restrict__ u_bf, const float* __restrict__ pos,
                      const int* __restrict__ src, const int* __restrict__ dst,
                      const __bf16* __restrict__ wt,
                      const float* __restrict__ ew1_l, const float* __restrict__ eb1_l,
                      const float* __restrict__ eb2_l, const float* __restrict__ gw_l,
                      const float* __restrict__ gb_l,  const float* __restrict__ pb1_l,
                      const float* __restrict__ pw2_l,
                      float* __restrict__ agg, float* __restrict__ possum) {
  __shared__ __bf16 wlds[WT_EDGE_ELEMS];        // 32KB: per-channel edge weights
  __shared__ __bf16 stage[8][16 * STG_LD];      // 18KB: per-wave D->A transpose staging
  const int wave = threadIdx.x >> 5, lane = threadIdx.x & 31;
  const int ln = lane & 15, hi = lane >> 4;
  const int c = blockIdx.y;
  const int tile = blockIdx.x * 8 + wave;       // E/16 tiles exactly

  // cooperative copy of this channel's edge weights into LDS (16B chunks)
  {
    const uint4* g = (const uint4*)(wt + WT_CH + (size_t)c * WT_CH_STR);
    uint4* s4 = (uint4*)wlds;
#pragma unroll
    for (int i = 0; i < WT_EDGE_ELEMS / 8 / 256; ++i)
      s4[threadIdx.x + i * 256] = g[threadIdx.x + i * 256];
  }
  __syncthreads();
  const __bf16* ew1s = wlds;
  const __bf16* ew1d = wlds + 4096;
  const __bf16* ew2t = wlds + 8192;
  const __bf16* pw1t = wlds + 12288;
  float* aggc = agg + (size_t)c * NN * DHID;

  const int e = tile * 16 + ln;                 // this lane's edge-row
  const int s = src[e];
  const int d = dst[e];
  const float rx = pos[s * 3 + 0] - pos[d * 3 + 0];
  const float ry = pos[s * 3 + 1] - pos[d * 3 + 1];
  const float rz = pos[s * 3 + 2] - pos[d * 3 + 2];
  const float d2 = rx * rx + ry * ry + rz * rz;

  const __bf16* rs = u_bf + (size_t)s * (NCH * DHID) + c * DHID;
  const __bf16* rd = u_bf + (size_t)d * (NCH * DHID) + c * DHID;
  v16bf as0 = afrag_bf(rs, 0), as1 = afrag_bf(rs, 32);
  v16bf ad0 = afrag_bf(rd, 0), ad1 = afrag_bf(rd, 32);

  // GEMM1: m1 = h_src @ W1[:64] + h_dst @ W1[64:128]
  v8f acc[4];
#pragma unroll
  for (int nt = 0; nt < 4; ++nt) {
    v8f a = {};
    a = wmma_bf(as0, bfragT(ew1s, 64, nt * 16, 0),  a);
    a = wmma_bf(as1, bfragT(ew1s, 64, nt * 16, 32), a);
    a = wmma_bf(ad0, bfragT(ew1d, 64, nt * 16, 0),  a);
    a = wmma_bf(ad1, bfragT(ew1d, 64, nt * 16, 32), a);
    acc[nt] = a;
  }

  // broadcast per-row (per-edge) scalars to the D-layout owners
  float d2m[8]; int dm[8];
#pragma unroll
  for (int v = 0; v < 8; ++v) { d2m[v] = __shfl(d2, 8 * hi + v); dm[v] = __shfl(d, 8 * hi + v); }

  __bf16* st = stage[wave];
  const float* eb1c  = eb1_l + c * DHID;
  const float* w1d2  = ew1_l + ((size_t)c * 129 + 128) * DHID;  // last row of edge_w1 (d2 term)
#pragma unroll
  for (int nt = 0; nt < 4; ++nt) {
    const int n = nt * 16 + ln;
    const float wr = w1d2[n], bb = eb1c[n];
#pragma unroll
    for (int v = 0; v < 8; ++v)
      st[(8 * hi + v) * STG_LD + n] = f2bf(siluf_(acc[nt][v] + d2m[v] * wr + bb));
  }

  // GEMM2: m2 = silu(m1) @ W2   (m1 re-read from LDS in A layout, already bf16)
  const __bf16* myrow = st + ln * STG_LD;
  v16bf a0 = afrag_bf(myrow, 0), a1 = afrag_bf(myrow, 32);
#pragma unroll
  for (int nt = 0; nt < 4; ++nt) {
    v8f a = {};
    a = wmma_bf(a0, bfragT(ew2t, 64, nt * 16, 0),  a);
    a = wmma_bf(a1, bfragT(ew2t, 64, nt * 16, 32), a);
    acc[nt] = a;
  }

  // bias + silu, then per-edge gate = sigmoid(m2 . gw + gb) via half-wave butterfly
  const float* eb2c = eb2_l + c * DHID;
  const float* gwc  = gw_l + c * DHID;
  const float  gb   = gb_l[c];
  float gw4[4], eb24[4];
#pragma unroll
  for (int nt = 0; nt < 4; ++nt) { gw4[nt] = gwc[nt * 16 + ln]; eb24[nt] = eb2c[nt * 16 + ln]; }
  float gv[8];
#pragma unroll
  for (int v = 0; v < 8; ++v) {
    float p = 0.0f;
#pragma unroll
    for (int nt = 0; nt < 4; ++nt) {
      float m2 = siluf_(acc[nt][v] + eb24[nt]);
      acc[nt][v] = m2;
      p += m2 * gw4[nt];
    }
#pragma unroll
    for (int off = 1; off < 16; off <<= 1) p += __shfl_xor(p, off);
    gv[v] = sigmoidf_(p + gb);
  }

  // gated message m3: scatter-add into agg[dst] + restage (bf16) for pos GEMM
#pragma unroll
  for (int nt = 0; nt < 4; ++nt) {
    const int n = nt * 16 + ln;
#pragma unroll
    for (int v = 0; v < 8; ++v) {
      const float m3 = acc[nt][v] * gv[v];
      atomicAdd(&aggc[(size_t)dm[v] * DHID + n], m3);
      st[(8 * hi + v) * STG_LD + n] = f2bf(m3);
    }
  }

  // GEMM3: q = m3 @ Pw1 ; p = silu(q + pb1) . pw2 (per edge)
  v16bf c0 = afrag_bf(myrow, 0), c1 = afrag_bf(myrow, 32);
  v8f q[4];
#pragma unroll
  for (int nt = 0; nt < 4; ++nt) {
    v8f a = {};
    a = wmma_bf(c0, bfragT(pw1t, 64, nt * 16, 0),  a);
    a = wmma_bf(c1, bfragT(pw1t, 64, nt * 16, 32), a);
    q[nt] = a;
  }
  const float* pb1c = pb1_l + c * DHID;
  const float* pw2c = pw2_l + c * DHID;
  float pb14[4], pw24[4];
#pragma unroll
  for (int nt = 0; nt < 4; ++nt) { pb14[nt] = pb1c[nt * 16 + ln]; pw24[nt] = pw2c[nt * 16 + ln]; }
  float pv[8];
#pragma unroll
  for (int v = 0; v < 8; ++v) {
    float p = 0.0f;
#pragma unroll
    for (int nt = 0; nt < 4; ++nt) p += siluf_(q[nt][v] + pb14[nt]) * pw24[nt];
#pragma unroll
    for (int off = 1; off < 16; off <<= 1) p += __shfl_xor(p, off);
    pv[v] = p;
  }

  // coordinate scatter: lanes ln<8 of each half handle edge m = 8*hi + ln
  const int msrc = 8 * hi + (ln & 7);
  const float rxm = __shfl(rx, msrc);
  const float rym = __shfl(ry, msrc);
  const float rzm = __shfl(rz, msrc);
  const int   ddm = __shfl(d, msrc);
  if (ln < 8) {
    float p = 0.0f;
#pragma unroll
    for (int v = 0; v < 8; ++v) p = (ln == v) ? pv[v] : p;   // static-index select
    const float tx = fminf(fmaxf(rxm * p, -CLAMP_V), CLAMP_V);
    const float ty = fminf(fmaxf(rym * p, -CLAMP_V), CLAMP_V);
    const float tz = fminf(fmaxf(rzm * p, -CLAMP_V), CLAMP_V);
    atomicAdd(&possum[ddm * 3 + 0], tx);
    atomicAdd(&possum[ddm * 3 + 1], ty);
    atomicAdd(&possum[ddm * 3 + 2], tz);
  }
}

// ---- node: v_bf[n, c*64+j] = bf16( [h | agg] @ nw + nb ) -----------------
__global__ __launch_bounds__(256)
void egnn_node_kernel(const __bf16* __restrict__ u_bf, const float* __restrict__ agg,
                      const __bf16* __restrict__ wt, const float* __restrict__ nb_l,
                      __bf16* __restrict__ v_bf) {
  const int wave = threadIdx.x >> 5, lane = threadIdx.x & 31;
  const int ln = lane & 15, hi = lane >> 4;
  const int c = blockIdx.y;
  const int mtile = blockIdx.x * 2 + (wave >> 2);
  const int nt = wave & 3;
  if (mtile >= (NN / 16)) return;
  const __bf16* nwt = wt + WT_CH + c * WT_CH_STR + WT_NW_OFF;   // [64][128]
  const __bf16* hrow = u_bf + (size_t)(mtile * 16 + ln) * (NCH * DHID) + c * DHID;
  const float*  arow = agg + ((size_t)c * NN + (mtile * 16 + ln)) * DHID;
  v8f a = {};
  a = wmma_bf(afrag_bf(hrow, 0),   bfragT(nwt, 128, nt * 16, 0),  a);
  a = wmma_bf(afrag_bf(hrow, 32),  bfragT(nwt, 128, nt * 16, 32), a);
  a = wmma_bf(afrag_f32(arow, 0),  bfragT(nwt, 128, nt * 16, 64), a);
  a = wmma_bf(afrag_f32(arow, 32), bfragT(nwt, 128, nt * 16, 96), a);
  const float bb = nb_l[c * DHID + nt * 16 + ln];
#pragma unroll
  for (int v = 0; v < 8; ++v)
    v_bf[(size_t)(mtile * 16 + 8 * hi + v) * (NCH * DHID) + c * DHID + nt * 16 + ln] = f2bf(a[v] + bb);
}

// ---- mix_out + residual: x += v_bf @ W[256,128] --------------------------
__global__ __launch_bounds__(256)
void egnn_mixout_kernel(const __bf16* __restrict__ v_bf, const __bf16* __restrict__ wt,
                        float* __restrict__ xio) {
  const int wave = threadIdx.x >> 5, lane = threadIdx.x & 31;
  const int ln = lane & 15, hi = lane >> 4;
  const int mtile = blockIdx.x;
  const int nt = wave;                          // 8 waves x 16 cols = 128
  const __bf16* mot = wt + WT_MIXOUT;           // [128][256]
  const __bf16* row = v_bf + (size_t)(mtile * 16 + ln) * (NCH * DHID);
  v8f a = {};
#pragma unroll
  for (int kk = 0; kk < 8; ++kk)
    a = wmma_bf(afrag_bf(row, kk * 32), bfragT(mot, 256, nt * 16, kk * 32), a);
#pragma unroll
  for (int v = 0; v < 8; ++v) {
    const size_t idx = (size_t)(mtile * 16 + 8 * hi + v) * D_IN + nt * 16 + ln;
    xio[idx] += a[v];
  }
}

// ---- host orchestration ---------------------------------------------------
extern "C" void kernel_launch(void* const* d_in, const int* in_sizes, int n_in,
                              void* d_out, int out_size, void* d_ws, size_t ws_size,
                              hipStream_t stream) {
  (void)in_sizes; (void)n_in; (void)out_size; (void)ws_size;
  const float* x_in   = (const float*)d_in[0];
  const float* pos_in = (const float*)d_in[1];
  const int*   ei     = (const int*)d_in[2];
  const float* mix_in_w  = (const float*)d_in[3];
  const float* mix_out_w = (const float*)d_in[4];
  const float* edge_w1 = (const float*)d_in[5];
  const float* edge_b1 = (const float*)d_in[6];
  const float* edge_w2 = (const float*)d_in[7];
  const float* edge_b2 = (const float*)d_in[8];
  const float* gate_w  = (const float*)d_in[9];
  const float* gate_b  = (const float*)d_in[10];
  const float* node_w  = (const float*)d_in[11];
  const float* node_b  = (const float*)d_in[12];
  const float* pos_w1  = (const float*)d_in[13];
  const float* pos_b1  = (const float*)d_in[14];
  const float* pos_w2  = (const float*)d_in[15];

  const int* src = ei;
  const int* dst = ei + EE;

  float* out_x   = (float*)d_out;
  float* out_pos = out_x + (size_t)NN * D_IN;

  uint8_t* p = (uint8_t*)d_ws;
  __bf16* u_bf = (__bf16*)p;  p += (size_t)NN * (NCH * DHID) * 2;
  __bf16* v_bf = (__bf16*)p;  p += (size_t)NN * (NCH * DHID) * 2;
  float* agg    = (float*)p;  p += (size_t)NCH * NN * DHID * 4;
  float* possum = (float*)p;  p += (((size_t)NN * 3 * 4) + 255) / 256 * 256;
  float* cntinv = (float*)p;  p += (size_t)NN * 4;
  __bf16* wtb   = (__bf16*)p; p += (size_t)WT_TOTAL * 2;

  // one-time: copy state into output buffers, build 1/(C*max(cnt,1))
  init_kernel<<<4096, 256, 0, stream>>>(x_in, pos_in, out_x, out_pos, cntinv);
  count_kernel<<<(EE + 255) / 256, 256, 0, stream>>>(dst, cntinv);
  inv_kernel<<<(NN + 255) / 256, 256, 0, stream>>>(cntinv);

  for (int l = 0; l < NLAY; ++l) {
    const float* mixin_l  = mix_in_w  + (size_t)l * D_IN * (NCH * DHID);
    const float* mixout_l = mix_out_w + (size_t)l * (NCH * DHID) * D_IN;
    const float* ew1_l = edge_w1 + (size_t)l * NCH * 129 * DHID;
    const float* eb1_l = edge_b1 + (size_t)l * NCH * DHID;
    const float* ew2_l = edge_w2 + (size_t)l * NCH * DHID * DHID;
    const float* eb2_l = edge_b2 + (size_t)l * NCH * DHID;
    const float* gw_l  = gate_w  + (size_t)l * NCH * DHID;
    const float* gb_l  = gate_b  + (size_t)l * NCH;
    const float* nw_l  = node_w  + (size_t)l * NCH * 2 * DHID * DHID;
    const float* nb_l  = node_b  + (size_t)l * NCH * DHID;
    const float* pw1_l = pos_w1  + (size_t)l * NCH * DHID * DHID;
    const float* pb1_l = pos_b1  + (size_t)l * NCH * DHID;
    const float* pw2_l = pos_w2  + (size_t)l * NCH * DHID;

    prep_weights_kernel<<<96, 256, 0, stream>>>(mixin_l, mixout_l, ew1_l, ew2_l, nw_l, pw1_l, wtb);
    egnn_mixin_kernel<<<NN / 16, 256, 0, stream>>>(out_x, wtb, u_bf);
    zero_kernel<<<4096, 256, 0, stream>>>(agg, possum);
    egnn_edge_kernel<<<dim3(EE / 16 / 8, NCH), 256, 0, stream>>>(
        u_bf, out_pos, src, dst, wtb, ew1_l, eb1_l, eb2_l, gw_l, gb_l, pb1_l, pw2_l, agg, possum);
    pos_update_kernel<<<(NN * 3 + 255) / 256, 256, 0, stream>>>(out_pos, possum, cntinv);
    egnn_node_kernel<<<dim3((NN / 16 + 1) / 2, NCH), 256, 0, stream>>>(u_bf, agg, wtb, nb_l, v_bf);
    egnn_mixout_kernel<<<NN / 16, 256, 0, stream>>>(v_bf, wtb, out_x);
  }
}
